// Theom_logit_12000138625194
// MI455X (gfx1250) — matrix operand order repaired and checked by reading, not compiled, not run on hardware
//
#include <hip/hip_runtime.h>

typedef __attribute__((ext_vector_type(16))) __bf16 v16bf;
typedef __attribute__((ext_vector_type(8)))  float  v8f;
typedef __attribute__((ext_vector_type(4)))  float  f32x4;
typedef __attribute__((ext_vector_type(4)))  unsigned int u32x4;
typedef __attribute__((ext_vector_type(8)))  int    i32x8;
typedef __attribute__((ext_vector_type(4)))  int    i32x4;

union Frag { v16bf bf; f32x4 q[2]; };
union BU   { __bf16 b; unsigned short u; };

#define EMBED 1024
#define H0    1024
#define H1C   512
#define NGOAL 128
#define NTHM  512
#define PM    32      // thm rows (pairs) per workgroup

#if defined(__HIP_DEVICE_COMPILE__) && defined(__gfx1250__) && \
    __has_builtin(__builtin_amdgcn_tensor_load_to_lds) && \
    __has_builtin(__builtin_amdgcn_s_wait_tensorcnt)
#define USE_TDM 1
#else
#define USE_TDM 0
#endif

// ---------------- prep: f32 -> bf16 convert ----------------
__global__ void cvt_f32_bf16(const float* __restrict__ src,
                             unsigned short* __restrict__ dst, int n) {
  int i = blockIdx.x * blockDim.x + threadIdx.x;
  if (i < n) {
    BU c; c.b = (__bf16)src[i];
    dst[i] = c.u;
  }
}

// W1mT[h][e] = W1[(2048+e)*1024 + h]  (bf16, transposed "mul" block of W1)
__global__ void make_w1mT(const float* __restrict__ W1,
                          unsigned short* __restrict__ W1mT) {
  int h = blockIdx.x;  // 0..1023
  for (int e = threadIdx.x; e < EMBED; e += blockDim.x) {
    BU c; c.b = (__bf16)W1[(size_t)(2 * EMBED + e) * H0 + h];
    W1mT[(size_t)h * EMBED + e] = c.u;
  }
}

// W2T[k][h] = W2[h*512 + k]  (bf16, transposed)
__global__ void make_w2T(const float* __restrict__ W2,
                         unsigned short* __restrict__ W2T) {
  int k = blockIdx.x;  // 0..511
  for (int h = threadIdx.x; h < H0; h += blockDim.x) {
    BU c; c.b = (__bf16)W2[(size_t)h * H1C + k];
    W2T[(size_t)k * H0 + h] = c.u;
  }
}

// G1[n][h] = b1[h] + sum_e g[n,e] * W1[e,h]        (rows 0..127)
// T1[m][h] =         sum_e t[m,e] * W1[1024+e,h]   (rows 128..639)
__global__ void precompute_gt(const float* __restrict__ g, const float* __restrict__ t,
                              const float* __restrict__ W1, const float* __restrict__ b1,
                              float* __restrict__ G1, float* __restrict__ T1) {
  int h = blockIdx.x * blockDim.x + threadIdx.x;   // grid.x=4, block=256 -> 0..1023
  int r = blockIdx.y;
  const float* src; const float* w; float* dst; float acc;
  if (r < NGOAL) {
    src = g + (size_t)r * EMBED; w = W1 + h;
    dst = G1 + (size_t)r * H0 + h; acc = b1[h];
  } else {
    int m = r - NGOAL;
    src = t + (size_t)m * EMBED; w = W1 + (size_t)EMBED * H0 + h;
    dst = T1 + (size_t)m * H0 + h; acc = 0.f;
  }
#pragma unroll 4
  for (int e = 0; e < EMBED; ++e) acc += src[e] * w[(size_t)e * H0];
  *dst = acc;
}

// ---------------- fused main kernel ----------------
// grid = (NTHM/PM, NGOAL), block = 256 (8 wave32)
// LDS: sG 2KB | sT 64KB | sH1 64KB | sH2 32KB  = 165888 B dynamic
__global__ __launch_bounds__(256) void fused_mlp(
    const unsigned short* __restrict__ g_bf,
    const unsigned short* __restrict__ t_bf,
    const unsigned short* __restrict__ w1mT,
    const unsigned short* __restrict__ w2T,
    const float* __restrict__ G1,
    const float* __restrict__ T1,
    const float* __restrict__ b2,
    const float* __restrict__ W3,
    const float* __restrict__ b3,
    float* __restrict__ out) {
  extern __shared__ char smem[];
  __bf16* sG  = (__bf16*)smem;                               // [1024]
  __bf16* sT  = (__bf16*)(smem + 2048);                      // [32][1024]
  __bf16* sH1 = (__bf16*)(smem + 2048 + 65536);              // [32][1024]
  __bf16* sH2 = (__bf16*)(smem + 2048 + 65536 + 65536);      // [32][512]

  const int tid  = threadIdx.x;
  const int w    = tid >> 5;
  const int lane = tid & 31;
  const int n    = blockIdx.y;
  const int m0   = blockIdx.x * PM;

  // ---- phase 0: stage bf16 g row + t tile into LDS ----
#if USE_TDM
  // Tensor Data Mover: DMA the contiguous 64KB t tile straight to LDS.
  if (w == 0) {
    const unsigned long long gaddr =
        (unsigned long long)(const void*)(t_bf + (size_t)m0 * EMBED);
    // D# group 0: count=1 | lds_addr | 57-bit global addr | type=2
    u32x4 g0;
    g0[0] = 1u;                                   // count=1, user mode
    g0[1] = 2048u;                                // LDS byte address of sT
    g0[2] = (unsigned int)(gaddr & 0xFFFFFFFFull);
    g0[3] = (unsigned int)((gaddr >> 32) & 0x1FFFFFFull) | (2u << 30);
    // D# group 1: data_size=8B, 1-D tile: tile_dim0 = 8192 elements
    const unsigned int ne = (unsigned int)(PM * EMBED / 4);  // 8192 x 8B
    i32x8 g1;
    g1[0] = (int)(3u << 16);                      // wg_mask=0, data_size=3 (8B)
    g1[1] = (int)((ne & 0xFFFFu) << 16);          // tensor_dim0 lo16 (<<16)
    g1[2] = (int)(((ne >> 16) & 0xFFFFu) | (1u << 16)); // dim0 hi16 | tensor_dim1=1 lo16
    g1[3] = (int)((ne & 0xFFFFu) << 16);          // tensor_dim1 hi16=0 | tile_dim0
    g1[4] = 0;                                    // tile_dim1=0, tile_dim2=0 (unused)
    g1[5] = (int)ne;                              // tensor_dim0_stride lo32
    g1[6] = 0;                                    // stride hi16 | dim1_stride lo16
    g1[7] = 0;                                    // dim1_stride hi32
    i32x4 zz4 = {0, 0, 0, 0};
    i32x8 zz8 = {0, 0, 0, 0, 0, 0, 0, 0};
    __builtin_amdgcn_tensor_load_to_lds(g0, g1, zz4, zz4, zz8, 0);
    // g row (2KB) by wave 0 lanes while TDM runs
    const f32x4* srcG = (const f32x4*)(g_bf + (size_t)n * EMBED);
    f32x4* dstG = (f32x4*)sG;
    for (int i = lane; i < EMBED / 8; i += 32) dstG[i] = srcG[i];
    __builtin_amdgcn_s_wait_tensorcnt(0);
  }
#else
  {
    const f32x4* srcT = (const f32x4*)(t_bf + (size_t)m0 * EMBED);
    f32x4* dstT = (f32x4*)sT;
    for (int i = tid; i < PM * EMBED / 8; i += 256) dstT[i] = srcT[i];
    const f32x4* srcG = (const f32x4*)(g_bf + (size_t)n * EMBED);
    f32x4* dstG = (f32x4*)sG;
    if (tid < EMBED / 8) dstG[tid] = srcG[tid];
  }
#endif
  __syncthreads();

  const int rb   = w & 1;                 // row-block of 16 pairs
  const int c16  = lane & 15;
  const int lrow = rb * 16 + c16;         // A-fragment source row (pair)
  const int b0   = (lane & 16) ? 8 : 0;   // A-layout K sub-offset
  const int bB   = (lane & 16) ? 16 : 0;  // unused; B loads use b0 too
  const int roff = (lane & 16) ? 8 : 0;   // C/D row offset

  // ---------- layer 1: H1 = relu(G1 + T1 + (g (*) t) @ W1m) ----------
  {
    const int hbase = (w >> 1) * 16;      // each wave owns 16 of 64 h-tiles
    const __bf16* bw = (const __bf16*)w1mT;
    for (int htg = 0; htg < 4; ++htg) {
      const int ht0 = hbase + htg * 4;
      v8f acc[4] = {};
      for (int k = 0; k < EMBED; k += 32) {
        Frag fg, ft, fa;
        fg.q[0] = *(const f32x4*)(sG + k + b0);
        fg.q[1] = *(const f32x4*)(sG + k + b0 + 16);
        ft.q[0] = *(const f32x4*)(sT + lrow * EMBED + k + b0);
        ft.q[1] = *(const f32x4*)(sT + lrow * EMBED + k + b0 + 16);
#pragma unroll
        for (int i = 0; i < 16; ++i)
          fa.bf[i] = (__bf16)((float)fg.bf[i] * (float)ft.bf[i]);
#pragma unroll
        for (int tt = 0; tt < 4; ++tt) {
          const int h = (ht0 + tt) * 16 + c16;
          Frag fb;
          const __bf16* bp = bw + (size_t)h * EMBED + k + b0;
          fb.q[0] = *(const f32x4*)bp;
          fb.q[1] = *(const f32x4*)(bp + 16);
          acc[tt] = __builtin_amdgcn_wmma_f32_16x16x32_bf16(
              false, fa.bf, false, fb.bf, (short)0, acc[tt], false, false);
        }
      }
#pragma unroll
      for (int tt = 0; tt < 4; ++tt) {
        const int h = (ht0 + tt) * 16 + c16;
        const float g1v = G1[(size_t)n * H0 + h];
#pragma unroll
        for (int j = 0; j < 8; ++j) {
          const int rg = rb * 16 + j + roff;
          float v = acc[tt][j] + g1v + T1[(size_t)(m0 + rg) * H0 + h];
          v = fmaxf(v, 0.f);
          sH1[rg * H0 + h] = (__bf16)v;
        }
      }
    }
  }
  __syncthreads();

  // ---------- layer 2: H2 = relu(H1 @ W2 + b2) ----------
  {
    const int cbase = (w >> 1) * 8;       // each wave owns 8 of 32 col-tiles
    const __bf16* bw = (const __bf16*)w2T;
    for (int ctg = 0; ctg < 2; ++ctg) {
      const int c0 = cbase + ctg * 4;
      v8f acc[4] = {};
      for (int k = 0; k < H0; k += 32) {
        Frag fa;
        fa.q[0] = *(const f32x4*)(sH1 + lrow * H0 + k + b0);
        fa.q[1] = *(const f32x4*)(sH1 + lrow * H0 + k + b0 + 16);
#pragma unroll
        for (int tt = 0; tt < 4; ++tt) {
          const int c = (c0 + tt) * 16 + c16;
          Frag fb;
          const __bf16* bp = bw + (size_t)c * H0 + k + b0;
          fb.q[0] = *(const f32x4*)bp;
          fb.q[1] = *(const f32x4*)(bp + 16);
          acc[tt] = __builtin_amdgcn_wmma_f32_16x16x32_bf16(
              false, fa.bf, false, fb.bf, (short)0, acc[tt], false, false);
        }
      }
#pragma unroll
      for (int tt = 0; tt < 4; ++tt) {
        const int c = (c0 + tt) * 16 + c16;
        const float bv = b2[c];
#pragma unroll
        for (int j = 0; j < 8; ++j) {
          const int rg = rb * 16 + j + roff;
          float v = acc[tt][j] + bv;
          v = fmaxf(v, 0.f);
          sH2[rg * H1C + c] = (__bf16)v;
        }
      }
    }
  }
  __syncthreads();

  // ---------- layer 3: out = H2 @ W3 + b3 (matvec, wave reduction) ----------
  {
    const float bias = b3[0];
    for (int p = w; p < PM; p += 8) {
      float a = 0.f;
#pragma unroll
      for (int j = 0; j < 16; ++j) {
        const int kk = lane + j * 32;
        a += (float)sH2[p * H1C + kk] * W3[kk];
      }
#pragma unroll
      for (int off = 16; off > 0; off >>= 1)
        a += __shfl_xor(a, off, 32);
      if (lane == 0) out[(size_t)n * NTHM + m0 + p] = a + bias;
    }
  }
}

// ---------------- host launcher ----------------
extern "C" void kernel_launch(void* const* d_in, const int* in_sizes, int n_in,
                              void* d_out, int out_size, void* d_ws, size_t ws_size,
                              hipStream_t stream) {
  const float* g  = (const float*)d_in[0];
  const float* t  = (const float*)d_in[1];
  const float* W1 = (const float*)d_in[2];
  const float* b1 = (const float*)d_in[3];
  const float* W2 = (const float*)d_in[4];
  const float* b2 = (const float*)d_in[5];
  const float* W3 = (const float*)d_in[6];
  const float* b3 = (const float*)d_in[7];
  float* out = (float*)d_out;

  char* ws = (char*)d_ws;
  size_t off = 0;
  auto alloc = [&](size_t bytes) -> char* {
    char* p = ws + off;
    off = (off + bytes + 255) & ~(size_t)255;
    return p;
  };
  float*          G1   = (float*)alloc((size_t)NGOAL * H0 * 4);
  float*          T1   = (float*)alloc((size_t)NTHM * H0 * 4);
  unsigned short* g_bf = (unsigned short*)alloc((size_t)NGOAL * EMBED * 2);
  unsigned short* t_bf = (unsigned short*)alloc((size_t)NTHM * EMBED * 2);
  unsigned short* w1mT = (unsigned short*)alloc((size_t)EMBED * H0 * 2);
  unsigned short* w2T  = (unsigned short*)alloc((size_t)H1C * H0 * 2);

  cvt_f32_bf16<<<(NGOAL * EMBED + 255) / 256, 256, 0, stream>>>(g, g_bf, NGOAL * EMBED);
  cvt_f32_bf16<<<(NTHM * EMBED + 255) / 256, 256, 0, stream>>>(t, t_bf, NTHM * EMBED);
  make_w1mT<<<H0, 256, 0, stream>>>(W1, w1mT);
  make_w2T<<<H1C, 256, 0, stream>>>(W2, w2T);
  precompute_gt<<<dim3(4, NGOAL + NTHM), 256, 0, stream>>>(g, t, W1, b1, G1, T1);

  const size_t shmem = 2048 + 65536 + 65536 + 32768;  // 165888 B (< 320KB/WGP)
  fused_mlp<<<dim3(NTHM / PM, NGOAL), 256, shmem, stream>>>(
      g_bf, t_bf, w1mT, w2T, G1, T1, b2, W3, b3, out);
}
